// GNNFourierFT_71605694759324
// MI455X (gfx1250) — matrix operand order repaired and checked by reading, not compiled, not run on hardware
//
#include <hip/hip_runtime.h>

#define N_NODES 10000
#define N_EDGES 160000
#define D_IN    512
#define D_OUT   256
#define HID     512
#define NCOEF   1000
#define ALPHA   1.0f
#define OUT_S   (N_NODES * D_OUT)   // 2,560,000 per output section

typedef __bf16 v16bf __attribute__((ext_vector_type(16)));
typedef float  v8f   __attribute__((ext_vector_type(8)));
typedef unsigned int u32x4 __attribute__((ext_vector_type(4)));

union Frag { v16bf v; u32x4 q[2]; };

// ---------------- bf16 split helpers (round-to-nearest-even) ----------------
__device__ __forceinline__ unsigned short f2bf(float f) {
    unsigned int u = __float_as_uint(f);
    u += 0x7FFFu + ((u >> 16) & 1u);
    return (unsigned short)(u >> 16);
}
__device__ __forceinline__ float bf2f(unsigned short h) {
    return __uint_as_float(((unsigned int)h) << 16);
}
__device__ __forceinline__ void split_bf(float f, unsigned short& hi, unsigned short& lo) {
    unsigned short h = f2bf(f);
    hi = h;
    lo = f2bf(f - bf2f(h));
}

// ---------------- elementwise utility kernels ----------------
__global__ void k_zero(float* __restrict__ p, int n) {
    int i = blockIdx.x * blockDim.x + threadIdx.x;
    if (i < n) p[i] = 0.0f;
}

__global__ void k_split(const float* __restrict__ src,
                        unsigned short* __restrict__ hi,
                        unsigned short* __restrict__ lo, int n) {
    int i = blockIdx.x * blockDim.x + threadIdx.x;
    if (i < n) { unsigned short h, l; split_bf(src[i], h, l); hi[i] = h; lo[i] = l; }
}

// relu(src + bias) -> bf16 hi/lo planes. bias may be null. colmask = cols-1 (pow2).
__global__ void k_relu_split(const float* __restrict__ src, const float* __restrict__ bias,
                             unsigned short* __restrict__ hi, unsigned short* __restrict__ lo,
                             int n, int colmask) {
    int i = blockIdx.x * blockDim.x + threadIdx.x;
    if (i >= n) return;
    float f = src[i] + (bias ? bias[i & colmask] : 0.0f);
    f = fmaxf(f, 0.0f);
    unsigned short h, l; split_bf(f, h, l);
    hi[i] = h; lo[i] = l;
}

// transpose + split: W[K][N] -> hi/lo[N][K]
__global__ void k_transpose_split(const float* __restrict__ W,
                                  unsigned short* __restrict__ hi,
                                  unsigned short* __restrict__ lo, int K, int N) {
    int i = blockIdx.x * blockDim.x + threadIdx.x;
    if (i >= K * N) return;
    int k = i / N, n = i % N;
    unsigned short h, l; split_bf(W[i], h, l);
    size_t t = (size_t)n * K + k;
    hi[t] = h; lo[t] = l;
}

// ---------------- graph degree / norm ----------------
__global__ void k_deg(const int* __restrict__ ei, float* __restrict__ deg, int E) {
    int e = blockIdx.x * blockDim.x + threadIdx.x;
    if (e < E) atomicAdd(&deg[ei[E + e]], 1.0f);   // dst = ei[1][e]
}

__global__ void k_dinv(const float* __restrict__ deg, float* __restrict__ dinv, int n) {
    int i = blockIdx.x * blockDim.x + threadIdx.x;
    if (i < n) dinv[i] = rsqrtf(fmaxf(deg[i], 1.0f));
}

// ---------------- sparse-spectrum iDFT: dW^T in bf16 hi/lo planes ----------------
// dW[m][n] = scale * sum_k c[k]*cos(2*pi*(m*r_k/R + n*c_k/C)); written transposed [n][m].
__global__ __launch_bounds__(256) void k_fourier_t(
    const int* __restrict__ Ef, const float* __restrict__ cf,
    unsigned short* __restrict__ bthi, unsigned short* __restrict__ btlo,
    int R, int C, float scale) {
    __shared__ int   sr[NCOEF];
    __shared__ int   sc[NCOEF];
    __shared__ float sv[NCOEF];
    for (int k = threadIdx.x; k < NCOEF; k += blockDim.x) {
        sr[k] = Ef[k]; sc[k] = Ef[NCOEF + k]; sv[k] = cf[k];
    }
    __syncthreads();
    int i = blockIdx.x * blockDim.x + threadIdx.x;
    if (i >= R * C) return;
    int m = i / C, n = i % C;
    float invR = 1.0f / (float)R, invC = 1.0f / (float)C;
    float sum = 0.0f;
    for (int k = 0; k < NCOEF; ++k) {
        int mr = (m * sr[k]) & (R - 1);             // R, C are powers of two
        int nc = (n * sc[k]) & (C - 1);
        float t = (float)mr * invR + (float)nc * invC;
        sum += sv[k] * __cosf(6.28318530717958647f * t);
    }
    float val = sum * scale;
    unsigned short h, l; split_bf(val, h, l);
    size_t ti = (size_t)n * R + m;
    bthi[ti] = h; btlo[ti] = l;
}

// ---------------- split-bf16 WMMA GEMM: C[M][N] = A[M][K] * Bt[N][K]^T ----------------
// grid.x = M/16 tiles, grid.y * 8 waves = N/16 tiles, 256 threads (8 wave32 waves).
__global__ __launch_bounds__(256) void k_gemm(
    const unsigned short* __restrict__ Ahi, const unsigned short* __restrict__ Alo,
    const unsigned short* __restrict__ Bthi, const unsigned short* __restrict__ Btlo,
    float* __restrict__ Cout, int N, int K) {
    const int lane = threadIdx.x & 31;
    const int wave = threadIdx.x >> 5;
    const int mt = blockIdx.x;
    const int nt = blockIdx.y * 8 + wave;
    const int r    = lane & 15;
    const int half = lane >> 4;
    const size_t aoff = (size_t)(mt * 16 + r) * K;
    const size_t boff = (size_t)(nt * 16 + r) * K;
    const unsigned short* arh = Ahi  + aoff;
    const unsigned short* arl = Alo  + aoff;
    const unsigned short* brh = Bthi + boff;
    const unsigned short* brl = Btlo + boff;
    const int oa = 16 * half;   // A: lane half picks contiguous K halves
    const int ob = 16 * half;   // B: lane half picks contiguous K halves

    v8f acc;
#pragma unroll
    for (int i = 0; i < 8; ++i) acc[i] = 0.0f;

    for (int k0 = 0; k0 < K; k0 += 32) {
        Frag ah, al, bh, bl;
        ah.q[0] = *(const u32x4*)(arh + k0 + oa);
        ah.q[1] = *(const u32x4*)(arh + k0 + oa + 8);
        al.q[0] = *(const u32x4*)(arl + k0 + oa);
        al.q[1] = *(const u32x4*)(arl + k0 + oa + 8);
        bh.q[0] = *(const u32x4*)(brh + k0 + ob);
        bh.q[1] = *(const u32x4*)(brh + k0 + ob + 8);
        bl.q[0] = *(const u32x4*)(brl + k0 + ob);
        bl.q[1] = *(const u32x4*)(brl + k0 + ob + 8);
        // fp32 ~= hi*hi + hi*lo + lo*hi, f32 accumulate
        acc = __builtin_amdgcn_wmma_f32_16x16x32_bf16(false, ah.v, false, bh.v, (short)0, acc, false, false);
        acc = __builtin_amdgcn_wmma_f32_16x16x32_bf16(false, ah.v, false, bl.v, (short)0, acc, false, false);
        acc = __builtin_amdgcn_wmma_f32_16x16x32_bf16(false, al.v, false, bh.v, (short)0, acc, false, false);
    }
    float* crow = Cout + (size_t)(mt * 16 + 8 * half) * N + (size_t)nt * 16 + r;
#pragma unroll
    for (int v = 0; v < 8; ++v) crow[(size_t)v * N] = acc[v];
}

// ---------------- edge scatter: acc[dst] += src_feat[src] * dinv[src]*dinv[dst] ----------------
__global__ __launch_bounds__(256) void k_scatter(
    const int* __restrict__ ei, const float* __restrict__ dinv,
    const float* __restrict__ src_feat, float* __restrict__ acc, int E, int D) {
    int e = blockIdx.x;
    int s = ei[e], d = ei[E + e];
    float coef = dinv[s] * dinv[d];
    const float* srow = src_feat + (size_t)s * D;
    float* arow = acc + (size_t)d * D;
    for (int c = threadIdx.x; c < D; c += blockDim.x)
        atomicAdd(&arow[c], srow[c] * coef);
}

// ---------------- finalize: emb_base, emb_total (emb_fourier already in out[2S..]) ----------------
__global__ void k_final(const float* __restrict__ acc2, const float* __restrict__ b1,
                        float* __restrict__ out, int n) {
    int i = blockIdx.x * blockDim.x + threadIdx.x;
    if (i >= n) return;
    float eb = acc2[i] + b1[i & (D_OUT - 1)];
    float ef = out[2 * n + i];
    out[n + i] = eb;       // emb_base
    out[i]     = eb + ef;  // emb_total
}

// ---------------- host launch ----------------
extern "C" void kernel_launch(void* const* d_in, const int* in_sizes, int n_in,
                              void* d_out, int out_size, void* d_ws, size_t ws_size,
                              hipStream_t stream) {
    const float* x   = (const float*)d_in[0];
    const int*   ei  = (const int*)  d_in[1];
    const float* W0  = (const float*)d_in[2];
    const float* b0  = (const float*)d_in[3];
    const float* W1  = (const float*)d_in[4];
    const float* b1  = (const float*)d_in[5];
    const int*   Ef0 = (const int*)  d_in[6];
    const float* cf0 = (const float*)d_in[7];
    const int*   Ef1 = (const int*)  d_in[8];
    const float* cf1 = (const float*)d_in[9];
    float* out = (float*)d_out;

    char* base = (char*)d_ws;
    size_t off = 0;
    auto alloc = [&](size_t bytes) -> char* {
        char* p = base + off;
        off = (off + bytes + 255) & ~(size_t)255;
        return p;
    };

    float* deg   = (float*)alloc(N_NODES * 4);
    float* dinv  = (float*)alloc(N_NODES * 4);
    unsigned short* xhi = (unsigned short*)alloc((size_t)N_NODES * D_IN * 2);
    unsigned short* xlo = (unsigned short*)alloc((size_t)N_NODES * D_IN * 2);
    unsigned short* w0thi  = (unsigned short*)alloc((size_t)HID * D_IN * 2);
    unsigned short* w0tlo  = (unsigned short*)alloc((size_t)HID * D_IN * 2);
    unsigned short* dw0thi = (unsigned short*)alloc((size_t)HID * D_IN * 2);
    unsigned short* dw0tlo = (unsigned short*)alloc((size_t)HID * D_IN * 2);
    unsigned short* w1thi  = (unsigned short*)alloc((size_t)D_OUT * HID * 2);
    unsigned short* w1tlo  = (unsigned short*)alloc((size_t)D_OUT * HID * 2);
    unsigned short* dw1thi = (unsigned short*)alloc((size_t)D_OUT * HID * 2);
    unsigned short* dw1tlo = (unsigned short*)alloc((size_t)D_OUT * HID * 2);
    float* bufA = (float*)alloc((size_t)N_NODES * HID * 4);   // xw0, later h hi/lo planes
    float* bufB = (float*)alloc((size_t)N_NODES * HID * 4);   // xfpre, later hw1 + acc2
    unsigned short* xfhi = (unsigned short*)alloc((size_t)N_NODES * HID * 2);
    unsigned short* xflo = (unsigned short*)alloc((size_t)N_NODES * HID * 2);
    float* bufD = (float*)alloc((size_t)N_NODES * HID * 4);   // acc1

    // reused regions
    unsigned short* hhi = (unsigned short*)bufA;                        // 10.24 MB
    unsigned short* hlo = (unsigned short*)bufA + (size_t)N_NODES * HID;
    float* hw1  = bufB;                                                 // 10.24 MB
    float* acc2 = bufB + (size_t)N_NODES * D_OUT;
    float* acc1 = bufD;

    const int T = 256;
    const int nXW = N_NODES * HID;   // 5,120,000
    const int nO  = OUT_S;           // 2,560,000

    // 1) degree -> dinv
    k_zero<<<(N_NODES + T - 1) / T, T, 0, stream>>>(deg, N_NODES);
    k_deg <<<(N_EDGES + T - 1) / T, T, 0, stream>>>(ei, deg, N_EDGES);
    k_dinv<<<(N_NODES + T - 1) / T, T, 0, stream>>>(deg, dinv, N_NODES);

    // 2) Fourier adapter weights (transposed bf16 planes), frozen weight transposes, x planes
    k_fourier_t<<<(D_IN * HID) / T, T, 0, stream>>>(Ef0, cf0, dw0thi, dw0tlo,
                                                    D_IN, HID, ALPHA / (float)(D_IN * HID));
    k_fourier_t<<<(HID * D_OUT) / T, T, 0, stream>>>(Ef1, cf1, dw1thi, dw1tlo,
                                                     HID, D_OUT, ALPHA / (float)(HID * D_OUT));
    k_transpose_split<<<(D_IN * HID) / T, T, 0, stream>>>(W0, w0thi, w0tlo, D_IN, HID);
    k_transpose_split<<<(HID * D_OUT) / T, T, 0, stream>>>(W1, w1thi, w1tlo, HID, D_OUT);
    k_split<<<(N_NODES * D_IN) / T, T, 0, stream>>>(x, xhi, xlo, N_NODES * D_IN);

    // 3) xw0 = x @ W0 ; xfpre = x @ dW0   (M=10000 -> 625 tiles, N=512 -> grid.y=4)
    k_gemm<<<dim3(N_NODES / 16, HID / 128), T, 0, stream>>>(xhi, xlo, w0thi, w0tlo, bufA, HID, D_IN);
    k_gemm<<<dim3(N_NODES / 16, HID / 128), T, 0, stream>>>(xhi, xlo, dw0thi, dw0tlo, bufB, HID, D_IN);

    // 4) xf = relu(xfpre) -> bf16 planes (frees bufB)
    k_relu_split<<<(nXW + T - 1) / T, T, 0, stream>>>(bufB, (const float*)nullptr,
                                                      xfhi, xflo, nXW, HID - 1);

    // 5) layer-1 aggregation: acc1[dst] += xw0[src]*coef ; h = relu(acc1 + b0) -> planes in bufA
    k_zero<<<(nXW + T - 1) / T, T, 0, stream>>>(acc1, nXW);
    k_scatter<<<N_EDGES, T, 0, stream>>>(ei, dinv, bufA, acc1, N_EDGES, HID);
    k_relu_split<<<(nXW + T - 1) / T, T, 0, stream>>>(acc1, b0, hhi, hlo, nXW, HID - 1);

    // 6) hw1 = h @ W1 (N=256 -> grid.y=2); layer-2 aggregation into acc2
    k_gemm<<<dim3(N_NODES / 16, D_OUT / 128), T, 0, stream>>>(hhi, hlo, w1thi, w1tlo, hw1, D_OUT, HID);
    k_zero<<<(nO + T - 1) / T, T, 0, stream>>>(acc2, nO);
    k_scatter<<<N_EDGES, T, 0, stream>>>(ei, dinv, hw1, acc2, N_EDGES, D_OUT);

    // 7) emb_fourier = xf @ dW1 -> out section 3 directly
    k_gemm<<<dim3(N_NODES / 16, D_OUT / 128), T, 0, stream>>>(xfhi, xflo, dw1thi, dw1tlo,
                                                              out + 2 * (size_t)nO, D_OUT, HID);

    // 8) emb_base = acc2 + b1 ; emb_total = emb_base + emb_fourier
    k_final<<<(nO + T - 1) / T, T, 0, stream>>>(acc2, b1, out, nO);
}
